// KernelRegressionAttention_16758962389757
// MI455X (gfx1250) — matrix operand order repaired and checked by reading, not compiled, not run on hardware
//
#include <hip/hip_runtime.h>

// ---------------------------------------------------------------------------
// KernelRegressionAttention for MI455X (gfx1250), wave32 + WMMA f16->f32.
// B=2, N=2048, D=2048, H=16, HD=128, CHUNK=64.
//
// Round-3 changes vs round-2:
//  * GEMM operands pre-converted in global memory: x -> f16 [M][K], each W ->
//    f16 transposed [N][K] (one 8MB buffer reused), attention output emitted
//    as f16. TDM tiles now land in fragment-ready f16 layout -> the in-loop
//    f32->f16 convert pass and its scattered ds_store_b16 are gone.
//  * GEMM K-step 32 -> 64, double-buffered TENSOR_LOAD_TO_LDS pipeline:
//    issue next tile pair, s_wait_tensorcnt 2 (in-order per wave), compute.
//  * Attention epilogue stages O through LDS and stores coalesced b128 f16.
// ---------------------------------------------------------------------------

typedef __attribute__((ext_vector_type(16))) _Float16 v16h;
typedef __attribute__((ext_vector_type(8)))  _Float16 v8h;
typedef __attribute__((ext_vector_type(4)))  _Float16 v4h;
typedef __attribute__((ext_vector_type(8)))  float    v8f;
typedef __attribute__((ext_vector_type(4)))  float    f32x4;
typedef __attribute__((ext_vector_type(4)))  unsigned int u32x4;
typedef __attribute__((ext_vector_type(8)))  int      i32x8;
typedef __attribute__((ext_vector_type(4)))  int      i32x4;

#define WMMA16(a, b, c) \
  __builtin_amdgcn_wmma_f32_16x16x32_f16(false, (a), false, (b), (short)0, (c), false, false)

// ---------------------------------------------------------------------------
// Fragment loaders. ISA 7.12.2 16-bit layouts:
//  A (16x32): element e at K = e + (e>=8 ? 8:0) + 8*half -> two 8-elem runs.
//  B (32x16): element e at K = e + (e>=8 ? 8:0) + 8*half -> two 8-elem runs.
// Storage: A row-major [m][k] f16; B stored TRANSPOSED [n][k] f16.
// Each fragment is exactly two aligned ds_load_b128.
// ---------------------------------------------------------------------------
__device__ __forceinline__ v16h frag_ldA(const _Float16* s, int row0, int k0,
                                         int ld, int lane) {
  int half = lane >> 4, m = row0 + (lane & 15);
  const _Float16* p = s + m * ld + k0 + 8 * half;
  v8h lo = *(const v8h*)p;
  v8h hi = *(const v8h*)(p + 16);
  return __builtin_shufflevector(lo, hi, 0, 1, 2, 3, 4, 5, 6, 7,
                                 8, 9, 10, 11, 12, 13, 14, 15);
}
__device__ __forceinline__ v16h frag_ldB(const _Float16* s, int k0, int n0,
                                         int ld, int lane) {
  int half = lane >> 4, n = n0 + (lane & 15);
  const _Float16* p = s + n * ld + k0 + 8 * half;
  v8h lo = *(const v8h*)p;
  v8h hi = *(const v8h*)(p + 16);
  return __builtin_shufflevector(lo, hi, 0, 1, 2, 3, 4, 5, 6, 7,
                                 8, 9, 10, 11, 12, 13, 14, 15);
}

// ---------------------------------------------------------------------------
// Tensor Data Mover 2D tile load (ISA ch. 8.3/8.4 descriptor layout).
// dsz_code: 0=1B, 1=2B, 2=4B elements. Guarded for toolchains w/o builtin.
// ---------------------------------------------------------------------------
#if __has_builtin(__builtin_amdgcn_tensor_load_to_lds)
#define HAS_TDM 1
#if __has_include(<hip/amd_detail/amd_gfx1250_TDM.h>)
#define TDM_CALL(g0, g1, g2, g3) \
  __builtin_amdgcn_tensor_load_to_lds((g0), (g1), (g2), (g3), \
                                      (i32x8){0, 0, 0, 0, 0, 0, 0, 0}, 0)
#else
#define TDM_CALL(g0, g1, g2, g3) \
  __builtin_amdgcn_tensor_load_to_lds((g0), (g1), (g2), (g3), 0)
#endif

__device__ __forceinline__ void tdm_load_2d(unsigned lds_addr, const void* gptr,
                                            unsigned dsz_code, unsigned tile_w,
                                            unsigned tile_h, unsigned pitch_elems) {
  unsigned long long ga = (unsigned long long)(__SIZE_TYPE__)gptr;
  u32x4 g0 = {0, 0, 0, 0};
  i32x8 g1 = {0, 0, 0, 0, 0, 0, 0, 0};
  i32x4 g2 = {0, 0, 0, 0};
  i32x4 g3 = {0, 0, 0, 0};
  // group0: count=1 | lds_addr | global_addr[56:0] | type=2 ("image")
  g0[0] = 1u;
  g0[1] = lds_addr;
  g0[2] = (unsigned)(ga & 0xFFFFFFFFu);
  g0[3] = (unsigned)((ga >> 32) & 0x01FFFFFFu) | (2u << 30);
  // group1: data_size; tensor_dim0/1 = tile dims (no OOB); stride0 = pitch.
  g1[0] = (int)(dsz_code << 16);
  g1[1] = (int)((tile_w & 0xFFFFu) << 16);                   // tensor_dim0 lo16
  g1[2] = (int)(((tile_w >> 16) & 0xFFFFu) | ((tile_h & 0xFFFFu) << 16));
  g1[3] = (int)(((tile_h >> 16) & 0xFFFFu) | ((tile_w & 0xFFFFu) << 16)); // tile_dim0
  g1[4] = (int)(tile_h & 0xFFFFu);                           // tile_dim1 (dim2=0)
  g1[5] = (int)pitch_elems;                                  // stride0 lo32
  g1[6] = 0;
  g1[7] = 0;
  TDM_CALL(g0, g1, g2, g3);
}
#endif  // HAS_TDM

// ---------------------------------------------------------------------------
// Elementwise f32 -> f16 (vectorized). n_quads = elems/4, one quad/thread.
// ---------------------------------------------------------------------------
__global__ __launch_bounds__(256) void cvt_f16_kernel(const float* __restrict__ in,
                                                      _Float16* __restrict__ out) {
  int i4 = blockIdx.x * 256 + threadIdx.x;
  f32x4 v = *(const f32x4*)&in[(__SIZE_TYPE__)i4 * 4];
  *(v4h*)&out[(__SIZE_TYPE__)i4 * 4] = __builtin_convertvector(v, v4h);
}

// ---------------------------------------------------------------------------
// Tiled transpose + f32 -> f16: outT[n][k] = (f16) in[k][n]. Dim x Dim.
// 64x64 tiles via padded LDS; coalesced loads and stores.
// ---------------------------------------------------------------------------
__global__ __launch_bounds__(256) void transpose_cvt_kernel(
    const float* __restrict__ in, _Float16* __restrict__ outT, int Dim) {
  __shared__ float tile[64 * 65];
  int r0 = blockIdx.y * 64, c0 = blockIdx.x * 64;
#pragma unroll
  for (int t = 0; t < 16; ++t) {
    int idx = threadIdx.x + 256 * t;
    int r = idx >> 6, c = idx & 63;
    tile[r * 65 + c] = in[(__SIZE_TYPE__)(r0 + r) * Dim + c0 + c];
  }
  __syncthreads();
#pragma unroll
  for (int t = 0; t < 16; ++t) {
    int idx = threadIdx.x + 256 * t;
    int rr = idx >> 6, cc = idx & 63;  // out row n=c0+rr, col k=r0+cc
    outT[(__SIZE_TYPE__)(c0 + rr) * Dim + r0 + cc] = (_Float16)tile[cc * 65 + rr];
  }
}

// ---------------------------------------------------------------------------
// GEMM: C[MxN] (f32) = A[MxK] @ B[KxN], f16 inputs: A row-major, B given
// TRANSPOSED as Bt[N][K]. 128x128 block tile, BK=64, 8 waves x (32x64).
// Double-buffered TDM staging; fragments are pure ds_load_b128.
// ---------------------------------------------------------------------------
__global__ __launch_bounds__(256) void gemm_f16(
    const _Float16* __restrict__ A, const _Float16* __restrict__ Bt,
    float* __restrict__ C, int M, int N, int K) {
  __shared__ _Float16 sA[2][128 * 64];  // [m][k]
  __shared__ _Float16 sB[2][128 * 64];  // [n][k]
  const int tid = threadIdx.x, lane = tid & 31, w = tid >> 5;
  const int wm = w & 3, wn = w >> 2;  // 4x2 wave grid -> 32x64 per wave
  const int m0 = blockIdx.y * 128, n0 = blockIdx.x * 128;
  const int half = lane >> 4, nl = lane & 15;
  const int nIter = K >> 6;

  v8f acc[2][4];
#pragma unroll
  for (int i = 0; i < 2; ++i)
#pragma unroll
    for (int j = 0; j < 4; ++j)
#pragma unroll
      for (int r = 0; r < 8; ++r) acc[i][j][r] = 0.f;

#if defined(HAS_TDM)
  if (w == 0) {
    tdm_load_2d((unsigned)(__SIZE_TYPE__)(void*)&sA[0][0],
                A + (__SIZE_TYPE__)m0 * K, 1u, 64u, 128u, (unsigned)K);
    tdm_load_2d((unsigned)(__SIZE_TYPE__)(void*)&sB[0][0],
                Bt + (__SIZE_TYPE__)n0 * K, 1u, 64u, 128u, (unsigned)K);
  }
#endif
  for (int it = 0; it < nIter; ++it) {
#if defined(HAS_TDM)
    const int buf = it & 1;
    if (w == 0) {
      if (it + 1 < nIter) {
        tdm_load_2d((unsigned)(__SIZE_TYPE__)(void*)&sA[buf ^ 1][0],
                    A + (__SIZE_TYPE__)m0 * K + (it + 1) * 64, 1u, 64u, 128u, (unsigned)K);
        tdm_load_2d((unsigned)(__SIZE_TYPE__)(void*)&sB[buf ^ 1][0],
                    Bt + (__SIZE_TYPE__)n0 * K + (it + 1) * 64, 1u, 64u, 128u, (unsigned)K);
        __builtin_amdgcn_s_wait_tensorcnt(2);  // current pair landed (in-order)
      } else {
        __builtin_amdgcn_s_wait_tensorcnt(0);
      }
    }
#else
    const int buf = 0;
#pragma unroll
    for (int t = 0; t < 4; ++t) {
      int i4 = tid + 256 * t;
      int r = i4 >> 3, c8 = i4 & 7;
      *(v8h*)&sA[0][r * 64 + c8 * 8] =
          *(const v8h*)&A[(__SIZE_TYPE__)(m0 + r) * K + it * 64 + c8 * 8];
      *(v8h*)&sB[0][r * 64 + c8 * 8] =
          *(const v8h*)&Bt[(__SIZE_TYPE__)(n0 + r) * K + it * 64 + c8 * 8];
    }
#endif
    __syncthreads();
    const _Float16* pA = &sA[buf][0];
    const _Float16* pB = &sB[buf][0];
#pragma unroll
    for (int k0 = 0; k0 < 64; k0 += 32) {
      v16h af[2], bf[4];
      af[0] = frag_ldA(pA, wm * 32, k0, 64, lane);
      af[1] = frag_ldA(pA, wm * 32 + 16, k0, 64, lane);
#pragma unroll
      for (int j = 0; j < 4; ++j) bf[j] = frag_ldB(pB, k0, wn * 64 + j * 16, 64, lane);
#pragma unroll
      for (int i = 0; i < 2; ++i)
#pragma unroll
        for (int j = 0; j < 4; ++j) acc[i][j] = WMMA16(af[i], bf[j], acc[i][j]);
    }
    __syncthreads();
  }
#pragma unroll
  for (int i = 0; i < 2; ++i)
#pragma unroll
    for (int j = 0; j < 4; ++j)
#pragma unroll
      for (int r = 0; r < 8; ++r) {
        int row = m0 + wm * 32 + i * 16 + r + 8 * half;
        int col = n0 + wn * 64 + j * 16 + nl;
        C[(__SIZE_TYPE__)row * N + col] = acc[i][j][r];
      }
}

// ---------------------------------------------------------------------------
// beta = 2*sigmoid(x @ Wb) — one wave per output, wave32 shfl reduction.
// ---------------------------------------------------------------------------
__global__ __launch_bounds__(256) void beta_kernel(
    const float* __restrict__ x, const float* __restrict__ Wb,
    float* __restrict__ beta, int rows) {
  int o = blockIdx.x * 8 + (threadIdx.x >> 5);
  int lane = threadIdx.x & 31;
  int row = o >> 4, h = o & 15;
  if (row >= rows) return;
  float s = 0.f;
  for (int d = lane; d < 2048; d += 32)
    s += x[(__SIZE_TYPE__)row * 2048 + d] * Wb[(__SIZE_TYPE__)d * 16 + h];
#pragma unroll
  for (int m = 16; m >= 1; m >>= 1) s += __shfl_xor(s, m, 32);
  if (lane == 0) beta[(__SIZE_TYPE__)row * 16 + h] = 2.f / (1.f + expf(-s));
}

// ---------------------------------------------------------------------------
// In-place RoPE (optional l2norm first) per (b,n,h) row of 128.
// ---------------------------------------------------------------------------
__global__ __launch_bounds__(256) void rope_kernel(float* __restrict__ a, int do_norm) {
  int o = blockIdx.x * 8 + (threadIdx.x >> 5);
  int lane = threadIdx.x & 31;
  int bn = o >> 4, h = o & 15;
  int n = bn & 2047;  // N = 2048
  float* p = a + (__SIZE_TYPE__)bn * 2048 + h * 128;
  float x0 = p[lane], x1 = p[lane + 32], x2 = p[lane + 64], x3 = p[lane + 96];
  if (do_norm) {
    float ss = x0 * x0 + x1 * x1 + x2 * x2 + x3 * x3;
#pragma unroll
    for (int m = 16; m >= 1; m >>= 1) ss += __shfl_xor(ss, m, 32);
    float r = rsqrtf(ss + 1e-6f);
    x0 *= r; x1 *= r; x2 *= r; x3 *= r;
  }
  const float LN_BASE_OVER_64 = 0.14391157f;  // ln(10000)/64
  float th0 = (float)n * expf(-(float)lane * LN_BASE_OVER_64);
  float th1 = (float)n * expf(-(float)(lane + 32) * LN_BASE_OVER_64);
  float c0, s0, c1, s1;
  __sincosf(th0, &s0, &c0);
  __sincosf(th1, &s1, &c1);
  p[lane]      = x0 * c0 - x2 * s0;
  p[lane + 64] = x0 * s0 + x2 * c0;
  p[lane + 32] = x1 * c1 - x3 * s1;
  p[lane + 96] = x1 * s1 + x3 * c1;
}

// ---------------------------------------------------------------------------
// DeltaNet chunked delta rule. One workgroup (8 waves) per (b,h).
// f32 masters: state S[128][128], Vc (solve), Am. f16 operands: Kc[t][d],
// KbT[dk][t], Sh[dv][dk] mirror, VnT[dv][t] mirror. ~192 KB WGP LDS.
// Writes Vn in place over v.
// ---------------------------------------------------------------------------
extern __shared__ __align__(16) char smem_raw[];

__global__ __launch_bounds__(256) void delta_kernel(
    const float* __restrict__ k, float* __restrict__ v, const float* __restrict__ beta) {
  float*    S   = (float*)smem_raw;                // 65536 B  [dk][dv]
  _Float16* Sh  = (_Float16*)(smem_raw + 65536);   // 32768 B  [dv][dk]
  _Float16* Kc  = (_Float16*)(smem_raw + 98304);   // 16384 B  [t][d]
  _Float16* KbT = (_Float16*)(smem_raw + 114688);  // 16384 B  [dk][t]
  float*    Vc  = (float*)(smem_raw + 131072);     // 32768 B  [t][dv]
  _Float16* VnT = (_Float16*)(smem_raw + 163840);  // 16384 B  [dv][t]
  float*    Am  = (float*)(smem_raw + 180224);     // 16384 B  [i][j]
  float*    Bt  = (float*)(smem_raw + 196608);     // 256 B
  const int tid = threadIdx.x, lane = tid & 31, w = tid >> 5;
  const int half = lane >> 4, nl = lane & 15;
  const int b = blockIdx.x >> 4, h = blockIdx.x & 15;
  const __SIZE_TYPE__ base = (__SIZE_TYPE__)b * 2048 * 2048 + (__SIZE_TYPE__)h * 128;

  for (int i = tid; i < 128 * 128; i += 256) S[i] = 0.f;
  for (int i = tid; i < 128 * 128; i += 256) Sh[i] = (_Float16)0.f;
  __syncthreads();

  for (int c = 0; c < 32; ++c) {
    if (tid < 64) Bt[tid] = beta[((__SIZE_TYPE__)b * 2048 + c * 64 + tid) * 16 + h];
    __syncthreads();
#pragma unroll
    for (int t = 0; t < 8; ++t) {
      int i4 = tid + 256 * t;
      int r = i4 >> 5, c4 = i4 & 31;
      __SIZE_TYPE__ g = base + (__SIZE_TYPE__)(c * 64 + r) * 2048 + c4 * 4;
      f32x4 kv = *(const f32x4*)&k[g];
      f32x4 vv = *(const f32x4*)&v[g];
      float bb = Bt[r];
      *(v4h*)&Kc[r * 128 + c4 * 4] = __builtin_convertvector(kv, v4h);
      *(f32x4*)&Vc[r * 128 + c4 * 4] = vv;
#pragma unroll
      for (int i = 0; i < 4; ++i) KbT[(c4 * 4 + i) * 64 + r] = (_Float16)(kv[i] * bb);
    }
    __syncthreads();

    // Am = strict_lower(K K^T) * beta[col]; 16 tiles, 2/wave.
    for (int tt = w * 2; tt < w * 2 + 2; ++tt) {
      int tm = tt >> 2, tn = tt & 3;
      v8f acc;
#pragma unroll
      for (int r = 0; r < 8; ++r) acc[r] = 0.f;
#pragma unroll
      for (int k0 = 0; k0 < 128; k0 += 32) {
        v16h a  = frag_ldA(Kc, tm * 16, k0, 128, lane);
        v16h bb = frag_ldB(Kc, k0, tn * 16, 128, lane);
        acc = WMMA16(a, bb, acc);
      }
#pragma unroll
      for (int r = 0; r < 8; ++r) {
        int gr = tm * 16 + r + 8 * half, gc = tn * 16 + nl;
        Am[gr * 64 + gc] = (gc < gr) ? acc[r] * Bt[gc] : 0.f;
      }
    }
    // Vc -= K @ S; 32 tiles, 4/wave.
    for (int tt = w * 4; tt < w * 4 + 4; ++tt) {
      int tm = tt >> 3, tn = tt & 7;
      v8f acc;
#pragma unroll
      for (int r = 0; r < 8; ++r) acc[r] = 0.f;
#pragma unroll
      for (int k0 = 0; k0 < 128; k0 += 32) {
        v16h a  = frag_ldA(Kc, tm * 16, k0, 128, lane);
        v16h bb = frag_ldB(Sh, k0, tn * 16, 128, lane);
        acc = WMMA16(a, bb, acc);
      }
#pragma unroll
      for (int r = 0; r < 8; ++r) {
        int gr = tm * 16 + r + 8 * half, gc = tn * 16 + nl;
        Vc[gr * 128 + gc] -= acc[r];
      }
    }
    __syncthreads();

    // forward substitution (I+Am) Vn = Vc; thread<128 owns a dv column.
    for (int i = 1; i < 64; ++i) {
      if (tid < 128) {
        float s = 0.f;
        for (int j = 0; j < i; ++j) s += Am[i * 64 + j] * Vc[j * 128 + tid];
        Vc[i * 128 + tid] -= s;
      }
      __syncthreads();
    }
    for (int i = tid; i < 64 * 128; i += 256) {
      int d = i >> 6, t = i & 63;
      VnT[i] = (_Float16)Vc[t * 128 + d];
    }
    __syncthreads();

    // S += (K*beta)^T @ Vn; 64 tiles, 8/wave; refresh Sh mirror.
    for (int tt = w * 8; tt < w * 8 + 8; ++tt) {
      int tm = tt >> 3, tn = tt & 7;
      v8f acc;
#pragma unroll
      for (int r = 0; r < 8; ++r) acc[r] = 0.f;
#pragma unroll
      for (int k0 = 0; k0 < 64; k0 += 32) {
        v16h a  = frag_ldA(KbT, tm * 16, k0, 64, lane);
        v16h bb = frag_ldB(VnT, k0, tn * 16, 64, lane);
        acc = WMMA16(a, bb, acc);
      }
#pragma unroll
      for (int r = 0; r < 8; ++r) {
        int gr = tm * 16 + r + 8 * half, gc = tn * 16 + nl;
        float nv = S[gr * 128 + gc] + acc[r];
        S[gr * 128 + gc] = nv;
        Sh[gc * 128 + gr] = (_Float16)nv;
      }
    }
#pragma unroll
    for (int t = 0; t < 8; ++t) {
      int i4 = tid + 256 * t;
      int r = i4 >> 5, c4 = i4 & 31;
      *(f32x4*)&v[base + (__SIZE_TYPE__)(c * 64 + r) * 2048 + c4 * 4] =
          *(const f32x4*)&Vc[r * 128 + c4 * 4];
    }
    __syncthreads();
  }
}

// ---------------------------------------------------------------------------
// Flash attention: S = (q*scale)@(k*beta)^T, causal online-softmax, O = P@v'.
// One workgroup per (qblock, b*h). Output written as f16 (for final GEMM),
// staged through LDS for coalesced b128 stores.
// ---------------------------------------------------------------------------
__global__ __launch_bounds__(256) void attn_kernel(
    const float* __restrict__ q, const float* __restrict__ k,
    const float* __restrict__ v, const float* __restrict__ beta,
    _Float16* __restrict__ o16) {
  _Float16* sQ  = (_Float16*)smem_raw;            // 16384 B [m][d]
  _Float16* sK  = (_Float16*)(smem_raw + 16384);  // 16384 B [t][d]
  _Float16* sVT = (_Float16*)(smem_raw + 32768);  // 16384 B [dv][t]
  float*    sP  = (float*)(smem_raw + 49152);     // 16384 B [m][t]
  _Float16* sPh = (_Float16*)(smem_raw + 65536);  //  8192 B [m][t]
  float*    sM  = (float*)(smem_raw + 73728);
  float*    sL  = (float*)(smem_raw + 73984);
  float*    sC  = (float*)(smem_raw + 74240);
  const int tid = threadIdx.x, lane = tid & 31, w = tid >> 5;
  const int half = lane >> 4, nl = lane & 15;
  const int qb = blockIdx.x;
  const int b = blockIdx.y >> 4, h = blockIdx.y & 15;
  const __SIZE_TYPE__ base = (__SIZE_TYPE__)b * 2048 * 2048 + (__SIZE_TYPE__)h * 128;
  const float scale = 0.08838834764831845f;  // 128^-0.5

#pragma unroll
  for (int t = 0; t < 8; ++t) {
    int i4 = tid + 256 * t;
    int r = i4 >> 5, c4 = i4 & 31;
    f32x4 qv = *(const f32x4*)&q[base + (__SIZE_TYPE__)(qb * 64 + r) * 2048 + c4 * 4];
    qv *= scale;
    *(v4h*)&sQ[r * 128 + c4 * 4] = __builtin_convertvector(qv, v4h);
  }
  if (tid < 64) { sM[tid] = -3.0e38f; sL[tid] = 0.f; }
  const int tmO = w >> 1, tnO = (w & 1) * 4;
  v8f accO[4];
#pragma unroll
  for (int t = 0; t < 4; ++t)
#pragma unroll
    for (int r = 0; r < 8; ++r) accO[t][r] = 0.f;
  __syncthreads();

  for (int kb = 0; kb <= qb; ++kb) {
#pragma unroll
    for (int t = 0; t < 8; ++t) {
      int i4 = tid + 256 * t;
      int r = i4 >> 5, c4 = i4 & 31;
      __SIZE_TYPE__ g = base + (__SIZE_TYPE__)(kb * 64 + r) * 2048 + c4 * 4;
      float bb = beta[((__SIZE_TYPE__)b * 2048 + kb * 64 + r) * 16 + h];
      f32x4 kv = *(const f32x4*)&k[g];
      f32x4 vv = *(const f32x4*)&v[g];
      kv *= bb;
      *(v4h*)&sK[r * 128 + c4 * 4] = __builtin_convertvector(kv, v4h);
#pragma unroll
      for (int i = 0; i < 4; ++i) sVT[(c4 * 4 + i) * 64 + r] = (_Float16)vv[i];
    }
    __syncthreads();
    // S = Q K'^T; 16 tiles, 2/wave.
    for (int tt = w * 2; tt < w * 2 + 2; ++tt) {
      int tm = tt >> 2, tn = tt & 3;
      v8f acc;
#pragma unroll
      for (int r = 0; r < 8; ++r) acc[r] = 0.f;
#pragma unroll
      for (int k0 = 0; k0 < 128; k0 += 32) {
        v16h a  = frag_ldA(sQ, tm * 16, k0, 128, lane);
        v16h bb = frag_ldB(sK, k0, tn * 16, 128, lane);
        acc = WMMA16(a, bb, acc);
      }
#pragma unroll
      for (int r = 0; r < 8; ++r) {
        int gr = tm * 16 + r + 8 * half, gc = tn * 16 + nl;
        sP[gr * 64 + gc] = (kb == qb && gc > gr) ? -3.0e38f : acc[r];
      }
    }
    __syncthreads();
    // online softmax; thread<64 owns a row; writes f16 prob mirror.
    if (tid < 64) {
      int row = tid;
      float mo = sM[row], mn = mo;
      for (int j = 0; j < 64; ++j) mn = fmaxf(mn, sP[row * 64 + j]);
      float corr = expf(mo - mn);
      float l = sL[row] * corr;
      for (int j = 0; j < 64; ++j) {
        float p = expf(sP[row * 64 + j] - mn);
        sPh[row * 64 + j] = (_Float16)p;
        l += p;
      }
      sM[row] = mn; sL[row] = l; sC[row] = corr;
    }
    __syncthreads();
    // rescale running O, then O += P @ V; 4 tiles/wave.
#pragma unroll
    for (int r = 0; r < 8; ++r) {
      float cc = sC[tmO * 16 + r + 8 * half];
#pragma unroll
      for (int t = 0; t < 4; ++t) accO[t][r] *= cc;
    }
#pragma unroll
    for (int t = 0; t < 4; ++t) {
      int tn = tnO + t;
#pragma unroll
      for (int k0 = 0; k0 < 64; k0 += 32) {
        v16h a  = frag_ldA(sPh, tmO * 16, k0, 64, lane);
        v16h bb = frag_ldB(sVT, k0, tn * 16, 64, lane);
        accO[t] = WMMA16(a, bb, accO[t]);
      }
    }
    __syncthreads();
  }
  // normalize, stage into sQ as f16, then coalesced b128 stores.
#pragma unroll
  for (int t = 0; t < 4; ++t) {
    int gcol = (tnO + t) * 16 + nl;
#pragma unroll
    for (int r = 0; r < 8; ++r) {
      int grow = tmO * 16 + r + 8 * half;
      sQ[grow * 128 + gcol] = (_Float16)(accO[t][r] / sL[grow]);
    }
  }
  __syncthreads();
#pragma unroll
  for (int t = 0; t < 4; ++t) {
    int i4 = tid + 256 * t;
    int r = i4 >> 4, c8 = i4 & 15;
    *(v8h*)&o16[base + (__SIZE_TYPE__)(qb * 64 + r) * 2048 + c8 * 8] =
        *(const v8h*)&sQ[r * 128 + c8 * 8];
  }
}

// ---------------------------------------------------------------------------
extern "C" void kernel_launch(void* const* d_in, const int* in_sizes, int n_in,
                              void* d_out, int out_size, void* d_ws, size_t ws_size,
                              hipStream_t stream) {
  const float* x  = (const float*)d_in[0];
  const float* Wq = (const float*)d_in[1];
  const float* Wk = (const float*)d_in[2];
  const float* Wv = (const float*)d_in[3];
  const float* Wo = (const float*)d_in[4];
  const float* Wb = (const float*)d_in[5];
  float* out = (float*)d_out;
  char*  ws  = (char*)d_ws;

  const int Bc = 2, Nn = 2048, Dd = 2048, Hh = 16;
  const int M = Bc * Nn;  // 4096

  float*    q    = (float*)(ws + 0);            // 32 MB
  float*    kk   = (float*)(ws + 33554432);     // 32 MB
  float*    vv   = (float*)(ws + 67108864);     // 32 MB (v_new in place)
  float*    bt   = (float*)(ws + 100663296);    // 256 KB
  _Float16* x16  = (_Float16*)(ws + 100925440); // 16 MB
  _Float16* wt16 = (_Float16*)(ws + 117702656); //  8 MB (reused per W)
  _Float16* o16  = (_Float16*)(ws + 126091264); // 16 MB
  // total ws use ~136.3 MB

  dim3 blk(256);
  dim3 g_gemm(Dd / 128, M / 128);   // (16, 32)
  dim3 g_tr(Dd / 64, Dd / 64);      // (32, 32)

  cvt_f16_kernel<<<(M * Dd) / 1024, blk, 0, stream>>>(x, x16);

  transpose_cvt_kernel<<<g_tr, blk, 0, stream>>>(Wq, wt16, Dd);
  gemm_f16<<<g_gemm, blk, 0, stream>>>(x16, wt16, q, M, Dd, Dd);
  transpose_cvt_kernel<<<g_tr, blk, 0, stream>>>(Wk, wt16, Dd);
  gemm_f16<<<g_gemm, blk, 0, stream>>>(x16, wt16, kk, M, Dd, Dd);
  transpose_cvt_kernel<<<g_tr, blk, 0, stream>>>(Wv, wt16, Dd);
  gemm_f16<<<g_gemm, blk, 0, stream>>>(x16, wt16, vv, M, Dd, Dd);

  beta_kernel<<<(M * Hh) / 8, blk, 0, stream>>>(x, Wb, bt, M);
  rope_kernel<<<(M * Hh) / 8, blk, 0, stream>>>(q, 0);   // rope(q)
  rope_kernel<<<(M * Hh) / 8, blk, 0, stream>>>(kk, 1);  // rope(l2norm(k))

  size_t delta_lds = 196864;  // S + Sh + Kc + KbT + Vc + VnT + Am + beta
  delta_kernel<<<Bc * Hh, blk, delta_lds, stream>>>(kk, vv, bt);

  size_t attn_lds = 74496;    // sQ + sK + sVT + sP + sPh + stats
  attn_kernel<<<dim3(Nn / 64, Bc * Hh), blk, attn_lds, stream>>>(q, kk, vv, bt, o16);

  transpose_cvt_kernel<<<g_tr, blk, 0, stream>>>(Wo, wt16, Dd);
  gemm_f16<<<g_gemm, blk, 0, stream>>>(o16, wt16, out, M, Dd, Dd);
}
